// TreeEncoder_90563680403627
// MI455X (gfx1250) — matrix operand order repaired and checked by reading, not compiled, not run on hardware
//
#include <hip/hip_runtime.h>
#include <hip/hip_bf16.h>

// ---------------------------------------------------------------------------
// TreeLSTM on MI455X (gfx1250): level-synchronous schedule, bf16 WMMA GEMMs.
//   Phase A: pack weights/embeddings to bf16, fold biases, build level lists.
//   Phase B: XALL[N,2048] = Xbf[N,512] @ WX^T        (v_wmma_f32_16x16x32_bf16)
//   Phase C: per level: ACC[M,2560] = HL@WL^T + HR@WR^T (gathered WMMA GEMM)
//            then pointwise LSTM cell -> out (h|c) + bf16 h for next level.
// ---------------------------------------------------------------------------

typedef __attribute__((ext_vector_type(16))) __bf16 v16bf;
typedef __attribute__((ext_vector_type(8)))  __bf16 v8bf;
typedef __attribute__((ext_vector_type(8)))  float  v8f;

#define HD    512
#define EMBD  512
#define XCOLS 2048   // 1536 (iou from x) + 512 (fx)
#define HCOLS 2560   // 1536 (iou from h) + 512 (fl part) + 512 (fr part)

__device__ __forceinline__ float sigm_(float x) { return 1.0f / (1.0f + __expf(-x)); }
__device__ __forceinline__ float tanh_(float x) { return 1.0f - 2.0f / (__expf(2.0f * x) + 1.0f); }

// ----------------------------- packing kernels -----------------------------

__global__ void __launch_bounds__(256) cvt_bf16_k(const float* __restrict__ s,
                                                  __bf16* __restrict__ d, int n) {
  int t = blockIdx.x * 256 + threadIdx.x;
  if (t < n) d[t] = (__bf16)s[t];
}

// Xbf[n][k] = bf16(E[vid[n]][k]); zero-pad rows >= N (GEMM row padding).
__global__ void __launch_bounds__(256) gather_x_k(const float* __restrict__ E,
                                                  const int* __restrict__ vid,
                                                  __bf16* __restrict__ Xbf,
                                                  int N, int rows) {
  int t = blockIdx.x * 256 + threadIdx.x;
  if (t >= rows * EMBD) return;
  int n = t >> 9, k = t & (EMBD - 1);
  float v = (n < N) ? E[(size_t)vid[n] * EMBD + k] : 0.0f;
  Xbf[t] = (__bf16)v;
}

// Folded biases + zero Hbf slot-0 row (the "no child" zero state).
__global__ void __launch_bounds__(256) bias_k(const float* __restrict__ biou_x,
                                              const float* __restrict__ biou_hl,
                                              const float* __restrict__ biou_hr,
                                              const float* __restrict__ bfx,
                                              const float* __restrict__ bfh,  // [4,512]
                                              float* __restrict__ bxiou,      // [1536]
                                              float* __restrict__ bfl,        // [512]
                                              float* __restrict__ bfr,        // [512]
                                              __bf16* __restrict__ Hbf) {
  int t = blockIdx.x * 256 + threadIdx.x;
  if (t < 3 * HD) bxiou[t] = biou_x[t] + biou_hl[t] + biou_hr[t];
  if (t < HD) {
    bfl[t] = bfx[t] + bfh[0 * HD + t] + bfh[1 * HD + t];
    bfr[t] = bfx[t] + bfh[2 * HD + t] + bfh[3 * HD + t];
    Hbf[t] = (__bf16)0.0f;  // slot 0 row
  }
}

// Sequential (postorder => children first): height per node + per-level lists.
__global__ void build_levels_k(const int* __restrict__ left, const int* __restrict__ right,
                               int N, int stride, int* __restrict__ height,
                               int* __restrict__ list) {
  if (threadIdx.x != 0 || blockIdx.x != 0) return;
  int cnt[32];
  for (int i = 0; i < 32; ++i) cnt[i] = 0;
  for (int i = 0; i < N; ++i) {
    int li = left[i], ri = right[i];
    int h = 0;
    if (li != 0) {
      int hl = height[li - 1], hr = height[ri - 1];
      h = (hl > hr ? hl : hr) + 1;
    }
    height[i] = h;
    list[h * stride + cnt[h]++] = i;
  }
}

// ------------------------------- GEMM kernels -------------------------------
// Wave-level 16x16 output tile, K in steps of 32 bf16.
// A (16x32 bf16) per-lane layout: lane<16 -> M=lane, K=k..k+15 contiguous;
//                                 lane>=16 -> M=lane-16, K=k+16..k+31.
// B (32x16 bf16) per-lane layout: lane<16 -> N=lane, K=k..k+15 contiguous;
//                                 lane>=16 -> N=lane-16, K=k+16..k+31.

__device__ __forceinline__ v16bf pack2(v8bf lo, v8bf hi) {
  v16bf r;
#pragma unroll
  for (int i = 0; i < 8; ++i) { r[i] = lo[i]; r[i + 8] = hi[i]; }
  return r;
}

// XALL[rows x 2048] = Xbf[rows x 512] @ WX^T   (WX is [2048 x 512] row-major)
__global__ void __launch_bounds__(256) gemm_x_k(const __bf16* __restrict__ Xbf,
                                                const __bf16* __restrict__ WX,
                                                float* __restrict__ XALL) {
  const int lane = threadIdx.x & 31;
  const int wave = threadIdx.x >> 5;
  const int row0 = blockIdx.x * 16;
  const int col0 = blockIdx.y * 128 + wave * 16;
  const int hi   = lane >> 4;
  const int l15  = lane & 15;

  const __bf16* ap = Xbf + (size_t)(row0 + l15) * EMBD + hi * 16;
  const __bf16* bp = WX  + (size_t)(col0 + l15) * EMBD + hi * 16;

  v8f acc = {};
#pragma unroll 4
  for (int k = 0; k < EMBD; k += 32) {
    v16bf a = pack2(*(const v8bf*)(ap + k), *(const v8bf*)(ap + k + 8));
    v16bf b = pack2(*(const v8bf*)(bp + k), *(const v8bf*)(bp + k + 8));
    acc = __builtin_amdgcn_wmma_f32_16x16x32_bf16(false, a, false, b, (short)0, acc,
                                                  false, false);
  }
  float* o = XALL + (size_t)(row0 + hi * 8) * XCOLS + col0 + l15;
#pragma unroll
  for (int v = 0; v < 8; ++v) o[(size_t)v * XCOLS] = acc[v];
}

// ACC[g,:2560] = Hbf[lslot(g)] @ WL^T + Hbf[rslot(g)] @ WR^T, gathered by level list.
__global__ void __launch_bounds__(256) gemm_h_k(const __bf16* __restrict__ Hbf,
                                                const __bf16* __restrict__ WL,
                                                const __bf16* __restrict__ WR,
                                                const int* __restrict__ list,
                                                const int* __restrict__ left,
                                                const int* __restrict__ right,
                                                float* __restrict__ ACC, int M) {
  const int lane = threadIdx.x & 31;
  const int wave = threadIdx.x >> 5;
  const int row0 = blockIdx.x * 16;
  const int col0 = blockIdx.y * 128 + wave * 16;
  const int hi   = lane >> 4;
  const int l15  = lane & 15;

  int pos = row0 + l15;
  int lslot = 0, rslot = 0;
  if (pos < M) {
    int n = list[pos];
    lslot = left[n];
    rslot = right[n];
  }
  const __bf16* apl = Hbf + (size_t)lslot * HD + hi * 16;
  const __bf16* apr = Hbf + (size_t)rslot * HD + hi * 16;
  const __bf16* bpl = WL + (size_t)(col0 + l15) * HD + hi * 16;
  const __bf16* bpr = WR + (size_t)(col0 + l15) * HD + hi * 16;

  v8f acc = {};
#pragma unroll 2
  for (int k = 0; k < HD; k += 32) {
    v16bf aL = pack2(*(const v8bf*)(apl + k), *(const v8bf*)(apl + k + 8));
    v16bf bL = pack2(*(const v8bf*)(bpl + k), *(const v8bf*)(bpl + k + 8));
    acc = __builtin_amdgcn_wmma_f32_16x16x32_bf16(false, aL, false, bL, (short)0, acc,
                                                  false, false);
    v16bf aR = pack2(*(const v8bf*)(apr + k), *(const v8bf*)(apr + k + 8));
    v16bf bR = pack2(*(const v8bf*)(bpr + k), *(const v8bf*)(bpr + k + 8));
    acc = __builtin_amdgcn_wmma_f32_16x16x32_bf16(false, aR, false, bR, (short)0, acc,
                                                  false, false);
  }
  float* o = ACC + (size_t)(row0 + hi * 8) * HCOLS + col0 + l15;
#pragma unroll
  for (int v = 0; v < 8; ++v) o[(size_t)v * HCOLS] = acc[v];
}

// ------------------------------ pointwise cell ------------------------------

__global__ void __launch_bounds__(256) cell_k(const float* __restrict__ XALL,
                                              const float* __restrict__ ACC,
                                              const float* __restrict__ bxiou,
                                              const float* __restrict__ bfl,
                                              const float* __restrict__ bfr,
                                              const int* __restrict__ list,
                                              const int* __restrict__ left,
                                              const int* __restrict__ right,
                                              float* __restrict__ out,
                                              __bf16* __restrict__ Hbf,
                                              int M, int useAcc) {
  int t = blockIdx.x * 256 + threadIdx.x;
  if (t >= M * HD) return;
  int g = t >> 9, j = t & (HD - 1);
  int n = list[g];
  int li = left[n], ri = right[n];
  const float* xr = XALL + (size_t)n * XCOLS;
  float aI = 0.f, aO = 0.f, aU = 0.f, aL = 0.f, aR = 0.f;
  if (useAcc) {
    const float* ar = ACC + (size_t)g * HCOLS;
    aI = ar[j]; aO = ar[HD + j]; aU = ar[2 * HD + j];
    aL = ar[3 * HD + j]; aR = ar[4 * HD + j];
  }
  float ig = sigm_(xr[j] + bxiou[j] + aI);
  float og = sigm_(xr[HD + j] + bxiou[HD + j] + aO);
  float ug = tanh_(xr[2 * HD + j] + bxiou[2 * HD + j] + aU);
  float fx = xr[3 * HD + j];
  float fl = sigm_(fx + bfl[j] + aL);
  float fr = sigm_(fx + bfr[j] + aR);
  float cl = li ? out[(size_t)(li - 1) * (2 * HD) + HD + j] : 0.0f;
  float cr = ri ? out[(size_t)(ri - 1) * (2 * HD) + HD + j] : 0.0f;
  float c = ig * ug + fl * cl + fr * cr;
  float h = og * tanh_(c);
  out[(size_t)n * (2 * HD) + j] = h;
  out[(size_t)n * (2 * HD) + HD + j] = c;
  Hbf[(size_t)(n + 1) * HD + j] = (__bf16)h;
}

// --------------------------------- launcher ---------------------------------

extern "C" void kernel_launch(void* const* d_in, const int* in_sizes, int n_in,
                              void* d_out, int out_size, void* d_ws, size_t ws_size,
                              hipStream_t stream) {
  const int*   vid     = (const int*)d_in[0];
  const int*   left    = (const int*)d_in[1];
  const int*   right   = (const int*)d_in[2];
  const float* E       = (const float*)d_in[3];
  const float* Wiou_x  = (const float*)d_in[4];
  const float* biou_x  = (const float*)d_in[5];
  const float* Wiou_hl = (const float*)d_in[6];
  const float* biou_hl = (const float*)d_in[7];
  const float* Wiou_hr = (const float*)d_in[8];
  const float* biou_hr = (const float*)d_in[9];
  const float* Wfx     = (const float*)d_in[10];
  const float* bfx     = (const float*)d_in[11];
  const float* Wfh     = (const float*)d_in[12];  // [4,512,512]
  const float* bfh     = (const float*)d_in[13];  // [4,512]
  float* out = (float*)d_out;

  const int N = in_sizes[0];                       // 4095
  int depth = 0;
  while (((1 << depth) - 1) < N) depth++;          // 12
  const int leafM   = (N + 1) >> 1;                // 2048 (also level-list stride)
  const int rows    = ((N + 15) / 16) * 16;        // 4096 padded GEMM rows
  const int hbfRows = (rows > N + 1) ? rows : (N + 1);
  const int maxM    = (N + 1) >> 2;                // 1024 max internal-level size
  const int accRows = ((maxM + 15) / 16) * 16;

  // Workspace carve-up (all from d_ws; ~60 MB total).
  size_t off = 0;
  auto carve = [&](size_t bytes) { size_t o = off; off = (off + bytes + 255) & ~(size_t)255; return o; };
  __bf16* Xbf   = (__bf16*)((char*)d_ws + carve((size_t)rows * EMBD * 2));
  __bf16* WX    = (__bf16*)((char*)d_ws + carve((size_t)XCOLS * EMBD * 2));
  __bf16* WL    = (__bf16*)((char*)d_ws + carve((size_t)HCOLS * HD * 2));
  __bf16* WR    = (__bf16*)((char*)d_ws + carve((size_t)HCOLS * HD * 2));
  __bf16* Hbf   = (__bf16*)((char*)d_ws + carve((size_t)hbfRows * HD * 2));
  float*  XALL  = (float*) ((char*)d_ws + carve((size_t)rows * XCOLS * 4));
  float*  ACC   = (float*) ((char*)d_ws + carve((size_t)accRows * HCOLS * 4));
  float*  bxiou = (float*) ((char*)d_ws + carve(3 * HD * 4));
  float*  bflv  = (float*) ((char*)d_ws + carve(HD * 4));
  float*  bfrv  = (float*) ((char*)d_ws + carve(HD * 4));
  int*    hts   = (int*)   ((char*)d_ws + carve((size_t)N * 4));
  int*    lvls  = (int*)   ((char*)d_ws + carve((size_t)depth * leafM * 4));

  const int WMAT = HD * HD;  // 512*512 elems
  auto cvt = [&](const float* s, __bf16* d, int n) {
    cvt_bf16_k<<<(n + 255) / 256, 256, 0, stream>>>(s, d, n);
  };

  // Phase A: pack bf16 weights (WX = [Wiou_x; Wfx], WL = [Wiou_hl; Wfh1; Wfh3],
  // WR = [Wiou_hr; Wfh0; Wfh2]), gather embeddings, fold biases, build levels.
  cvt(Wiou_x,        WX,                3 * HD * EMBD);
  cvt(Wfx,           WX + 3 * HD * EMBD, WMAT);
  cvt(Wiou_hl,       WL,                3 * HD * HD);
  cvt(Wfh + 1 * WMAT, WL + 3 * HD * HD,  WMAT);
  cvt(Wfh + 3 * WMAT, WL + 4 * HD * HD,  WMAT);
  cvt(Wiou_hr,       WR,                3 * HD * HD);
  cvt(Wfh + 0 * WMAT, WR + 3 * HD * HD,  WMAT);
  cvt(Wfh + 2 * WMAT, WR + 4 * HD * HD,  WMAT);
  gather_x_k<<<(rows * EMBD + 255) / 256, 256, 0, stream>>>(E, vid, Xbf, N, rows);
  bias_k<<<(3 * HD + 255) / 256, 256, 0, stream>>>(biou_x, biou_hl, biou_hr, bfx, bfh,
                                                   bxiou, bflv, bfrv, Hbf);
  build_levels_k<<<1, 1, 0, stream>>>(left, right, N, leafM, hts, lvls);

  // Phase B: x-path GEMM for all nodes at once.
  {
    dim3 g(rows / 16, XCOLS / 128);
    gemm_x_k<<<g, 256, 0, stream>>>(Xbf, WX, XALL);
  }

  // Phase C: level-by-level recurrence (leaves -> root).
  for (int lvl = 0; lvl < depth; ++lvl) {
    int M = (N + 1) >> (lvl + 1);      // nodes at this level (full tree)
    const int* list = lvls + (size_t)lvl * leafM;
    if (lvl > 0) {
      dim3 g((M + 15) / 16, HCOLS / 128);
      gemm_h_k<<<g, 256, 0, stream>>>(Hbf, WL, WR, list, left, right, ACC, M);
    }
    cell_k<<<(M * HD + 255) / 256, 256, 0, stream>>>(XALL, ACC, bxiou, bflv, bfrv,
                                                     list, left, right, out, Hbf,
                                                     M, lvl > 0 ? 1 : 0);
  }
}
